// BPTransformer_23983097381106
// MI455X (gfx1250) — matrix operand order repaired and checked by reading, not compiled, not run on hardware
//
#include <hip/hip_runtime.h>
#include <hip/hip_bf16.h>
#include <math.h>

// ---------------------------------------------------------------------------
// Types for WMMA fragments (gfx1250, wave32)
// ---------------------------------------------------------------------------
typedef __attribute__((ext_vector_type(16))) _Float16 v16h;
typedef __attribute__((ext_vector_type(8)))  _Float16 v8h;
typedef __attribute__((ext_vector_type(8)))  float    v8f;

__device__ __forceinline__ v8f wmma32(v16h a, v16h b, v8f c) {
    // D = A(16x32 f16) * B(32x16 f16) + C(16x16 f32)
    return __builtin_amdgcn_wmma_f32_16x16x32_f16(
        /*neg_a=*/false, a, /*neg_b=*/false, b,
        /*c_mod=*/(short)0, c, /*reuse_a=*/false, /*reuse_b=*/false);
}

// A-matrix fragment (16x32, MxK). Lane L<16 holds row M=L: halves 0..7 = K0..7,
// halves 8..15 = K16..23. Lane L>=16: K8..15 and K24..31 of row M=L-16.
__device__ __forceinline__ v16h frag_a_lds(const _Float16* rowBase, int lane) {
    union { v16h v; v8h h[2]; } u;
    const int lo = (lane < 16) ? 0 : 8;
    u.h[0] = *(const v8h*)(rowBase + lo);
    u.h[1] = *(const v8h*)(rowBase + lo + 16);
    return u.v;
}

// B-matrix fragment (32x16, KxN). Lane L<16 holds column N=L: halves 0..7 = K0..7,
// halves 8..15 = K16..23. Lane L>=16: K8..15 and K24..31 of column N=L-16.
__device__ __forceinline__ v16h frag_b_lds(const _Float16* colBase, int lane) {
    union { v16h v; v8h h[2]; } u;
    const int off = (lane < 16) ? 0 : 8;
    u.h[0] = *(const v8h*)(colBase + off);
    u.h[1] = *(const v8h*)(colBase + off + 16);
    return u.v;
}

// Pack 8 f32 (two float4) -> 8 f16
__device__ __forceinline__ v8h cvt8(float4 a, float4 b) {
    v8h h;
    h[0] = (_Float16)a.x; h[1] = (_Float16)a.y; h[2] = (_Float16)a.z; h[3] = (_Float16)a.w;
    h[4] = (_Float16)b.x; h[5] = (_Float16)b.y; h[6] = (_Float16)b.z; h[7] = (_Float16)b.w;
    return h;
}

// ---------------------------------------------------------------------------
// Problem constants
// ---------------------------------------------------------------------------
#define BB 4
#define NDET 128
#define NT 2048
#define NYX 256
#define DD 256
#define LL 6
#define NTOK 256
#define IMG_PIX (BB * NYX * NYX)   // 262144

// ---------------------------------------------------------------------------
// 0) f32 -> f16 weight conversion (once per launch)
// ---------------------------------------------------------------------------
__global__ void __launch_bounds__(256)
cvt_f16_kernel(const float* __restrict__ src, _Float16* __restrict__ dst, int n)
{
    const int i = (blockIdx.x * 256 + threadIdx.x) * 8;
    if (i >= n) return;
    const float4 a = *(const float4*)(src + i);
    const float4 b = *(const float4*)(src + i + 4);
    *(v8h*)(dst + i) = cvt8(a, b);
}

// ---------------------------------------------------------------------------
// 1) Backprojection: one wave per pixel, 4 detectors/lane, shfl reduction.
// ---------------------------------------------------------------------------
__global__ void __launch_bounds__(256)
backproject_kernel(const float* __restrict__ sino, const float* __restrict__ lut,
                   float* __restrict__ bp, float* __restrict__ bp_out)
{
    const int tid  = threadIdx.x;
    const int w    = tid >> 5;
    const int lane = tid & 31;
    const int pix  = blockIdx.x * 8 + w;
    const int b    = pix >> 16;
    const int yx   = pix & 65535;

    const float2* lut2 = (const float2*)lut;
    const float*  sb   = sino + (size_t)b * NDET * NT;

    float acc = 0.f, asum = 0.f;
#pragma unroll
    for (int i = 0; i < 4; ++i) {
        const int d = lane + i * 32;
        const float2 ka = lut2[(size_t)yx * NDET + d];
        const int   kf    = (int)ka.x;
        const float alpha = ka.y;
        const bool  valid = (kf >= 0) && (kf < NT - 1);
        const int   k0    = kf < 0 ? 0 : (kf > NT - 2 ? NT - 2 : kf);
        const float s0 = sb[(size_t)d * NT + k0];
        const float s1 = sb[(size_t)d * NT + k0 + 1];
        const float apod = 0.5f - 0.5f * __cosf(6.283185307179586f * (float)d / 127.0f);
        asum += apod;
        acc  += valid ? apod * ((1.f - alpha) * s0 + alpha * s1) : 0.f;
    }
#pragma unroll
    for (int ofs = 16; ofs; ofs >>= 1) {
        acc  += __shfl_xor(acc,  ofs);
        asum += __shfl_xor(asum, ofs);
    }
    if (lane == 0) {
        const float r = acc / fmaxf(asum, 1e-6f);
        bp[pix]     = r;
        bp_out[pix] = r;
    }
}

// ---------------------------------------------------------------------------
// 2) Patchify: bp (B,256,256) -> xp (B*N, 256)
// ---------------------------------------------------------------------------
__global__ void __launch_bounds__(256)
patchify_kernel(const float* __restrict__ bp, float* __restrict__ xp)
{
    const int p   = blockIdx.x * 256 + threadIdx.x;
    const int b   = p >> 16;
    const int rem = p & 65535;
    const int tok = rem >> 8;
    const int pel = rem & 255;
    const int y = (tok >> 4) * 16 + (pel >> 4);
    const int x = (tok & 15) * 16 + (pel & 15);
    xp[p] = bp[(size_t)b * 65536 + y * 256 + x];
}

// ---------------------------------------------------------------------------
// 3) WMMA GEMM (templated shapes): out(1024xN) = A(1024xK,f32) @ W16(NxK)^T + bias
//    Double-buffered LDS pipeline: next tile's async B load + A convert are
//    issued before the current tile's WMMAs; one barrier per K-step.
//    MODE: 0 store | 1 +pos_embed | 2 out+= | 3 gelu(exact)
// ---------------------------------------------------------------------------
template<int N, int K, int MODE>
__global__ void __launch_bounds__(256)
wmma_gemm_kernel(const float* __restrict__ A, const _Float16* __restrict__ W16,
                 const float* __restrict__ bias, float* __restrict__ out,
                 const float* __restrict__ extra)
{
    constexpr int NK = K / 32;
    __shared__ __align__(16) _Float16 As[2][64 * 40];   // row stride 40 halves
    __shared__ __align__(16) _Float16 Bs[2][64 * 40];

    const int tid  = threadIdx.x;
    const int w    = tid >> 5;
    const int lane = tid & 31;
    const int m0   = blockIdx.x * 64;
    const int n0   = blockIdx.y * 64;
    const int mSub = w & 3;
    const int nSub = (w >> 2) * 2;

    // Per-thread staging assignment: row = tid/4, 8-elem chunk = (tid%4)*8
    const int srow = tid >> 2;
    const int scol = (tid & 3) * 8;
    const float*    aSrc = A   + (size_t)(m0 + srow) * K + scol;
    const _Float16* bSrc = W16 + (size_t)(n0 + srow) * K + scol;
    _Float16* aDst0 = &As[0][srow * 40 + scol];
    _Float16* aDst1 = &As[1][srow * 40 + scol];
    // Low 32 bits of a generic LDS pointer == wave-relative LDS byte offset.
    const uint32_t bOff0 = (uint32_t)(uintptr_t)&Bs[0][srow * 40 + scol];
    const uint32_t bOff1 = (uint32_t)(uintptr_t)&Bs[1][srow * 40 + scol];

    // Prologue: stage tile 0 into buffer 0
    asm volatile("global_load_async_to_lds_b128 %0, %1, off"
                 :: "v"(bOff0), "v"(bSrc) : "memory");
    *(v8h*)aDst0 = cvt8(*(const float4*)aSrc, *(const float4*)(aSrc + 4));
    aSrc += 32; bSrc += 32;
    asm volatile("s_wait_asynccnt 0x0" ::: "memory");
    __syncthreads();

    v8f c0 = {}; v8f c1 = {};
#pragma unroll 2
    for (int it = 0; it < NK; ++it) {
        const int cur = it & 1;
        if (it + 1 < NK) {   // prefetch next tile into the other buffer
            const uint32_t bOff = cur ? bOff0 : bOff1;
            _Float16*      aDst = cur ? aDst0 : aDst1;
            asm volatile("global_load_async_to_lds_b128 %0, %1, off"
                         :: "v"(bOff), "v"(bSrc) : "memory");
            *(v8h*)aDst = cvt8(*(const float4*)aSrc, *(const float4*)(aSrc + 4));
            aSrc += 32; bSrc += 32;
        }
        const _Float16* Ab = As[cur];
        const _Float16* Bb = Bs[cur];
        const v16h a  = frag_a_lds(&Ab[(mSub * 16 + (lane & 15)) * 40], lane);
        const v16h b0 = frag_b_lds(&Bb[((nSub + 0) * 16 + (lane & 15)) * 40], lane);
        const v16h b1 = frag_b_lds(&Bb[((nSub + 1) * 16 + (lane & 15)) * 40], lane);
        c0 = wmma32(a, b0, c0);
        c1 = wmma32(a, b1, c1);
        asm volatile("s_wait_asynccnt 0x0" ::: "memory");
        __syncthreads();
    }

    const int col0  = n0 + nSub * 16 + (lane & 15);
    const int col1  = col0 + 16;
    const int rBase = m0 + mSub * 16 + ((lane >> 4) << 3);
    const float bv0 = bias[col0];
    const float bv1 = bias[col1];
#pragma unroll
    for (int r = 0; r < 8; ++r) {
        const int row = rBase + r;
        const float v0 = c0[r] + bv0;
        const float v1 = c1[r] + bv1;
        float* p0 = out + (size_t)row * N + col0;
        float* p1 = out + (size_t)row * N + col1;
        if (MODE == 0) { *p0 = v0; *p1 = v1; }
        else if (MODE == 1) {
            const int t = row & 255;
            *p0 = v0 + extra[(size_t)t * N + col0];
            *p1 = v1 + extra[(size_t)t * N + col1];
        } else if (MODE == 2) { *p0 += v0; *p1 += v1; }
        else {
            *p0 = 0.5f * v0 * (1.0f + erff(v0 * 0.70710678118654752f));
            *p1 = 0.5f * v1 * (1.0f + erff(v1 * 0.70710678118654752f));
        }
    }
}

// ---------------------------------------------------------------------------
// 4) LayerNorm over D=256: one block per row
// ---------------------------------------------------------------------------
__global__ void __launch_bounds__(256)
layernorm_kernel(const float* __restrict__ x, const float* __restrict__ g,
                 const float* __restrict__ bta, float* __restrict__ y)
{
    __shared__ float red[8], red2[8];
    const int row = blockIdx.x, tid = threadIdx.x;
    const int w = tid >> 5, lane = tid & 31;
    const float v = x[(size_t)row * 256 + tid];
    float s = v, sq = v * v;
#pragma unroll
    for (int ofs = 16; ofs; ofs >>= 1) { s += __shfl_xor(s, ofs); sq += __shfl_xor(sq, ofs); }
    if (lane == 0) { red[w] = s; red2[w] = sq; }
    __syncthreads();
    float ts = 0.f, tsq = 0.f;
#pragma unroll
    for (int i = 0; i < 8; ++i) { ts += red[i]; tsq += red2[i]; }
    const float mean = ts * (1.0f / 256.0f);
    const float var  = tsq * (1.0f / 256.0f) - mean * mean;
    const float rstd = rsqrtf(var + 1e-5f);
    y[(size_t)row * 256 + tid] = (v - mean) * rstd * g[tid] + bta[tid];
}

// ---------------------------------------------------------------------------
// 5) Attention: one block per (b, h, 32-query tile). WMMA QK^T and PV.
// ---------------------------------------------------------------------------
__global__ void __launch_bounds__(256)
attention_kernel(const float* __restrict__ qkv, float* __restrict__ o)
{
    __shared__ __align__(16) float    Ssm[32 * 256];   // 32KB, reused as f16 Vt
    __shared__ __align__(16) _Float16 Ks[256 * 40];    // 20KB, reused as f16 P
    __shared__ __align__(16) _Float16 Qs[32 * 40];

    const int tid  = threadIdx.x;
    const int w    = tid >> 5;
    const int lane = tid & 31;
    const int qt = blockIdx.x & 7;
    const int h  = (blockIdx.x >> 3) & 7;
    const int b  = blockIdx.x >> 6;
    const size_t base = (size_t)b * NTOK * (3 * DD);

    // Q tile (32x32): 128 chunks of 8 floats
    if (tid < 128) {
        const int r = tid >> 2, c8 = (tid & 3) * 8;
        const float* s = &qkv[base + (size_t)(qt * 32 + r) * 768 + h * 32 + c8];
        *(v8h*)&Qs[r * 40 + c8] = cvt8(*(const float4*)s, *(const float4*)(s + 4));
    }
    // K rows (256x32): 1024 chunks of 8 floats
    for (int e = tid; e < 1024; e += 256) {
        const int r = e >> 2, c8 = (e & 3) * 8;
        const float* s = &qkv[base + (size_t)r * 768 + 256 + h * 32 + c8];
        *(v8h*)&Ks[r * 40 + c8] = cvt8(*(const float4*)s, *(const float4*)(s + 4));
    }
    __syncthreads();

    // S = (Q K^T) * 1/sqrt(32)
    {
        const int mS = w & 1;
        const int nG = (w >> 1) * 4;
        const v16h a = frag_a_lds(&Qs[(mS * 16 + (lane & 15)) * 40], lane);
#pragma unroll
        for (int j = 0; j < 4; ++j) {
            const int nS = nG + j;
            const v16h bf = frag_b_lds(&Ks[(nS * 16 + (lane & 15)) * 40], lane);
            v8f c = {};
            c = wmma32(a, bf, c);
            const int colS = nS * 16 + (lane & 15);
            const int rowB = mS * 16 + ((lane >> 4) << 3);
#pragma unroll
            for (int r = 0; r < 8; ++r)
                Ssm[(rowB + r) * 256 + colS] = c[r] * 0.17677669529663687f;
        }
    }
    __syncthreads();

    // Row softmax -> f16 P in (dead) Ks region
    _Float16* P = Ks;
    for (int i = 0; i < 4; ++i) {
        const int r = w * 4 + i;
        float mx = -1e30f;
#pragma unroll
        for (int ct = 0; ct < 8; ++ct) mx = fmaxf(mx, Ssm[r * 256 + lane + ct * 32]);
#pragma unroll
        for (int ofs = 16; ofs; ofs >>= 1) mx = fmaxf(mx, __shfl_xor(mx, ofs));
        float s = 0.f;
#pragma unroll
        for (int ct = 0; ct < 8; ++ct) {
            const int idx = r * 256 + lane + ct * 32;
            const float e = __expf(Ssm[idx] - mx);
            Ssm[idx] = e;
            s += e;
        }
#pragma unroll
        for (int ofs = 16; ofs; ofs >>= 1) s += __shfl_xor(s, ofs);
        const float inv = 1.0f / s;
#pragma unroll
        for (int ct = 0; ct < 8; ++ct) {
            const int idx = r * 256 + lane + ct * 32;
            P[idx] = (_Float16)(Ssm[idx] * inv);
        }
    }
    __syncthreads();

    // V^T (32 x 256 tokens) into (dead) S region (transpose gather)
    _Float16* Vt = (_Float16*)Ssm;
    for (int e = tid; e < 32 * 256; e += 256) {
        const int n = e >> 8, t = e & 255;
        Vt[n * 256 + t] = (_Float16)qkv[base + (size_t)t * 768 + 512 + h * 32 + n];
    }
    __syncthreads();

    // O(32x32) = P(32x256) @ V(256x32)
    if (w < 4) {
        const int mS = w & 1, nS = w >> 1;
        v8f c = {};
#pragma unroll
        for (int k = 0; k < 8; ++k) {
            const v16h a  = frag_a_lds(&P [(mS * 16 + (lane & 15)) * 256 + k * 32], lane);
            const v16h bf = frag_b_lds(&Vt[(nS * 16 + (lane & 15)) * 256 + k * 32], lane);
            c = wmma32(a, bf, c);
        }
        const int col  = h * 32 + nS * 16 + (lane & 15);
        const int rowB = qt * 32 + mS * 16 + ((lane >> 4) << 3);
#pragma unroll
        for (int r = 0; r < 8; ++r)
            o[((size_t)b * 256 + rowB + r) * 256 + col] = c[r];
    }
}

// ---------------------------------------------------------------------------
// 6) Depatchify + 3x3 SAME conv + bias + bp residual -> d_out (img)
// ---------------------------------------------------------------------------
__global__ void __launch_bounds__(256)
conv_out_kernel(const float* __restrict__ pixels, const float* __restrict__ cw,
                const float* __restrict__ cb, const float* __restrict__ bp,
                float* __restrict__ outImg)
{
    const int p   = blockIdx.x * 256 + threadIdx.x;
    const int b   = p >> 16;
    const int rem = p & 65535;
    const int y = rem >> 8, x = rem & 255;
    float acc = cb[0] + bp[p];
#pragma unroll
    for (int ky = 0; ky < 3; ++ky) {
        const int yy = y + ky - 1;
        if (yy < 0 || yy > 255) continue;
#pragma unroll
        for (int kx = 0; kx < 3; ++kx) {
            const int xx = x + kx - 1;
            if (xx < 0 || xx > 255) continue;
            const int row = b * 256 + (yy >> 4) * 16 + (xx >> 4);
            const int col = (yy & 15) * 16 + (xx & 15);
            acc += cw[ky * 3 + kx] * pixels[(size_t)row * 256 + col];
        }
    }
    outImg[p] = acc;
}

// ---------------------------------------------------------------------------
// Host-side orchestration
// ---------------------------------------------------------------------------
extern "C" void kernel_launch(void* const* d_in, const int* in_sizes, int n_in,
                              void* d_out, int out_size, void* d_ws, size_t ws_size,
                              hipStream_t stream)
{
    (void)in_sizes; (void)n_in; (void)out_size; (void)ws_size;
    const float* sino    = (const float*)d_in[0];
    const float* lut     = (const float*)d_in[1];
    const float* patch_w = (const float*)d_in[2];
    const float* patch_b = (const float*)d_in[3];
    const float* pos     = (const float*)d_in[4];
    const float* ln1_g   = (const float*)d_in[5];
    const float* ln1_b   = (const float*)d_in[6];
    const float* wqkv    = (const float*)d_in[7];
    const float* bqkv    = (const float*)d_in[8];
    const float* wo      = (const float*)d_in[9];
    const float* bo      = (const float*)d_in[10];
    const float* ln2_g   = (const float*)d_in[11];
    const float* ln2_b   = (const float*)d_in[12];
    const float* w1      = (const float*)d_in[13];
    const float* b1      = (const float*)d_in[14];
    const float* w2      = (const float*)d_in[15];
    const float* b2      = (const float*)d_in[16];
    const float* proj_w  = (const float*)d_in[17];
    const float* proj_b  = (const float*)d_in[18];
    const float* conv_w  = (const float*)d_in[19];
    const float* conv_b  = (const float*)d_in[20];
    float* out = (float*)d_out;   // [img | bp_img]

    // f32 scratch
    float* ws   = (float*)d_ws;
    float* bp   = ws; ws += IMG_PIX;
    float* xp   = ws; ws += IMG_PIX;
    float* z    = ws; ws += IMG_PIX;
    float* ybuf = ws; ws += IMG_PIX;
    float* qkv  = ws; ws += 1024 * 768;
    float* obuf = ws; ws += IMG_PIX;
    float* hbuf = ws; ws += 1024 * 1024;
    float* pix  = ws; ws += IMG_PIX;
    // f16 weight scratch (converted once per launch)
    _Float16* wh = (_Float16*)ws;
    _Float16* patch_w16 = wh; wh += 256 * 256;
    _Float16* wqkv16    = wh; wh += LL * 768 * 256;
    _Float16* wo16      = wh; wh += LL * 256 * 256;
    _Float16* w1_16     = wh; wh += LL * 1024 * 256;
    _Float16* w2_16     = wh; wh += LL * 256 * 1024;
    _Float16* proj_w16  = wh; wh += 256 * 256;

    cvt_f16_kernel<<<(256*256)/2048,     256, 0, stream>>>(patch_w, patch_w16, 256*256);
    cvt_f16_kernel<<<(LL*768*256)/2048,  256, 0, stream>>>(wqkv,    wqkv16,    LL*768*256);
    cvt_f16_kernel<<<(LL*256*256)/2048,  256, 0, stream>>>(wo,      wo16,      LL*256*256);
    cvt_f16_kernel<<<(LL*1024*256)/2048, 256, 0, stream>>>(w1,      w1_16,     LL*1024*256);
    cvt_f16_kernel<<<(LL*256*1024)/2048, 256, 0, stream>>>(w2,      w2_16,     LL*256*1024);
    cvt_f16_kernel<<<(256*256)/2048,     256, 0, stream>>>(proj_w,  proj_w16,  256*256);

    backproject_kernel<<<IMG_PIX / 8, 256, 0, stream>>>(sino, lut, bp, out + IMG_PIX);
    patchify_kernel<<<IMG_PIX / 256, 256, 0, stream>>>(bp, xp);
    wmma_gemm_kernel<256, 256, 1><<<dim3(16, 4), 256, 0, stream>>>(xp, patch_w16, patch_b, z, pos);
    for (int l = 0; l < LL; ++l) {
        layernorm_kernel<<<1024, 256, 0, stream>>>(z, ln1_g + l * 256, ln1_b + l * 256, ybuf);
        wmma_gemm_kernel<768, 256, 0><<<dim3(16, 12), 256, 0, stream>>>(
            ybuf, wqkv16 + (size_t)l * 768 * 256, bqkv + l * 768, qkv, nullptr);
        attention_kernel<<<256, 256, 0, stream>>>(qkv, obuf);
        wmma_gemm_kernel<256, 256, 2><<<dim3(16, 4), 256, 0, stream>>>(
            obuf, wo16 + (size_t)l * 256 * 256, bo + l * 256, z, nullptr);
        layernorm_kernel<<<1024, 256, 0, stream>>>(z, ln2_g + l * 256, ln2_b + l * 256, ybuf);
        wmma_gemm_kernel<1024, 256, 3><<<dim3(16, 16), 256, 0, stream>>>(
            ybuf, w1_16 + (size_t)l * 1024 * 256, b1 + l * 1024, hbuf, nullptr);
        wmma_gemm_kernel<256, 1024, 2><<<dim3(16, 4), 256, 0, stream>>>(
            hbuf, w2_16 + (size_t)l * 256 * 1024, b2 + l * 256, z, nullptr);
    }
    wmma_gemm_kernel<256, 256, 0><<<dim3(16, 4), 256, 0, stream>>>(z, proj_w16, proj_b, pix, nullptr);
    conv_out_kernel<<<IMG_PIX / 256, 256, 0, stream>>>(pix, conv_w, conv_b, bp, out);
}
